// GCN_15307263443205
// MI455X (gfx1250) — matrix operand order repaired and checked by reading, not compiled, not run on hardware
//
#include <hip/hip_runtime.h>
#include <hip/hip_bf16.h>

// ---------------------------------------------------------------------------
// GCN (3x GCNConv + FC) for MI455X / gfx1250.
// GEMMs: v_wmma_f32_16x16x32_bf16, fp32 accumulate.
//   - Weights pre-packed once into bf16 WMMA fragment layout -> B fragment is
//     a single contiguous 32B vector load per tile (L2/L0-hot).
//   - One wave per 16-row M-strip holds NT accumulators and reuses the A
//     fragment across all N-tiles (A read exactly once).
// Edge aggregation: global_atomic_add_f32 on the 19.2MB (L2-resident) agg.
// ---------------------------------------------------------------------------

typedef __attribute__((ext_vector_type(16))) __bf16 v16bf;
typedef __attribute__((ext_vector_type(8)))  float  v8f;

#define GCN_NODES 50000
#define GCN_EDGES 800000
#define GCN_INDIM 512
#define GCN_HID   96
#define GCN_OUT   64
#define GCN_MT    (GCN_NODES / 16)   // 3125 M-tiles

__device__ __forceinline__ __bf16 f2bf(float f) { return (__bf16)f; }

// ---------------------------------------------------------------------------
__global__ void gcn_zero_f32(float* __restrict__ p, int n) {
  int i = blockIdx.x * blockDim.x + threadIdx.x;
  if (i < n) p[i] = 0.0f;
}

__global__ void gcn_deg_count(const int* __restrict__ dst, float* __restrict__ deg, int E) {
  int i = blockIdx.x * blockDim.x + threadIdx.x;
  if (i < E) atomicAdd(&deg[dst[i]], 1.0f);
}

__global__ void gcn_make_dinv(float* __restrict__ deg, int n) {
  int i = blockIdx.x * blockDim.x + threadIdx.x;
  if (i < n) deg[i] = rsqrtf(deg[i] + 1.0f);
}

// ---------------------------------------------------------------------------
// Pack W[K,N] (fp32, row-major) into bf16 WMMA B-fragment layout:
//   P[((kt*ntN + nt)*32 + lane)*16 + e] = bf16( W[(kt*32 + (lane>>4)*8 + (e&7) + (e>>3)*16)*N
//                                                 + nt*16 + (lane&15)] )
// so a wave's B fragment for tile (kt,nt) is 16 contiguous bf16 per lane.
// ---------------------------------------------------------------------------
__global__ void gcn_pack_w(const float* __restrict__ W, __bf16* __restrict__ P,
                           int K, int N) {
  int idx = blockIdx.x * blockDim.x + threadIdx.x;
  if (idx >= K * N) return;
  int e    = idx & 15;
  int lane = (idx >> 4) & 31;
  int t    = idx >> 9;              // tile index = kt*ntN + nt
  int ntN  = N >> 4;
  int kt   = t / ntN;
  int nt   = t - kt * ntN;
  int k    = kt * 32 + (lane >> 4) * 8 + (e & 7) + (e >> 3) * 16;
  int n    = nt * 16 + (lane & 15);
  P[idx] = f2bf(W[(size_t)k * N + n]);
}

// ---------------------------------------------------------------------------
// C[M, NT*16] = A[M, K] @ Bpacked (+ optional bias).
// One wave per 16-row M-strip; NT accumulators; A fragment loaded once per
// K-step and reused by NT back-to-back WMMAs.
// A-fragment (16x32 bf16): lane row = lane%16; half-wave K groups:
//   e0..7 -> K = kb + half*8 + 0..7, e8..15 -> K = kb + 16 + half*8 + 0..7
// C/D (16x16 f32): element v -> row M = 8*half + v, col N = lane%16
// ---------------------------------------------------------------------------
template <int K, int NT>
__global__ void gcn_gemm_wmma(const float* __restrict__ A,
                              const __bf16* __restrict__ Bp,
                              const float* __restrict__ bias,
                              float* __restrict__ C, int Mtiles) {
  constexpr int N = NT * 16;
  int wave = (blockIdx.x * blockDim.x + threadIdx.x) >> 5;
  int lane = threadIdx.x & 31;
  if (wave >= Mtiles) return;        // wave-uniform: EXEC all-1s for WMMA

  int half = lane >> 4;
  int l15  = lane & 15;
  int m    = wave * 16 + l15;

  v8f zero = {};
  v8f acc[NT];
#pragma unroll
  for (int t = 0; t < NT; ++t) acc[t] = zero;

  const float* Arow  = A + (size_t)m * K + half * 8;
  const v16bf* Bfrag = (const v16bf*)Bp;

  for (int kt = 0; kt < K / 32; ++kt) {
    // ---- A tile: 4 coalesced float4 loads, convert to bf16 in registers ----
    float4 a0 = *(const float4*)(Arow + 0);
    float4 a1 = *(const float4*)(Arow + 4);
    float4 a2 = *(const float4*)(Arow + 16);
    float4 a3 = *(const float4*)(Arow + 20);
    Arow += 32;
    v16bf af;
    af[0]  = f2bf(a0.x); af[1]  = f2bf(a0.y); af[2]  = f2bf(a0.z); af[3]  = f2bf(a0.w);
    af[4]  = f2bf(a1.x); af[5]  = f2bf(a1.y); af[6]  = f2bf(a1.z); af[7]  = f2bf(a1.w);
    af[8]  = f2bf(a2.x); af[9]  = f2bf(a2.y); af[10] = f2bf(a2.z); af[11] = f2bf(a2.w);
    af[12] = f2bf(a3.x); af[13] = f2bf(a3.y); af[14] = f2bf(a3.z); af[15] = f2bf(a3.w);

#pragma unroll
    for (int t = 0; t < NT; ++t) {
      // single contiguous 32B fragment load (2x global_load_b128), L0/L2-hot
      v16bf bfv = Bfrag[(size_t)(kt * NT + t) * 32 + lane];
      acc[t] = __builtin_amdgcn_wmma_f32_16x16x32_bf16(
          /*neg_a=*/false, af, /*neg_b=*/false, bfv,
          /*c_mod=*/(short)0, acc[t], /*reuse_a=*/false, /*reuse_b=*/false);
    }
  }

  int rbase = wave * 16 + half * 8;
#pragma unroll
  for (int t = 0; t < NT; ++t) {
    float bv = bias ? bias[t * 16 + l15] : 0.0f;
#pragma unroll
    for (int v = 0; v < 8; ++v) {
      C[(size_t)(rbase + v) * N + t * 16 + l15] = acc[t][v] + bv;
    }
  }
}

// ---------------------------------------------------------------------------
// One wave per edge; lane covers features {lane, lane+32, lane+64} (HID=96).
// agg[dst] += h[src] * dinv[src]*dinv[dst]  via global_atomic_add_f32.
// ---------------------------------------------------------------------------
__global__ void gcn_scatter_edges(const int* __restrict__ src,
                                  const int* __restrict__ dst,
                                  const float* __restrict__ dinv,
                                  const float* __restrict__ h,
                                  float* __restrict__ agg, int E) {
  int wave = (blockIdx.x * blockDim.x + threadIdx.x) >> 5;
  int lane = threadIdx.x & 31;
  if (wave >= E) return;
  int   s = src[wave];
  int   d = dst[wave];
  float w = dinv[s] * dinv[d];
  const float* hp = h + (size_t)s * GCN_HID;
  float*       ap = agg + (size_t)d * GCN_HID;
  atomicAdd(&ap[lane],      hp[lane]      * w);
  atomicAdd(&ap[lane + 32], hp[lane + 32] * w);
  atomicAdd(&ap[lane + 64], hp[lane + 64] * w);
}

// ---------------------------------------------------------------------------
// In-place: agg = relu(agg + h*dinv^2 + b)  -> becomes next layer activation.
// ---------------------------------------------------------------------------
__global__ void gcn_finalize(float* __restrict__ agg,
                             const float* __restrict__ h,
                             const float* __restrict__ dinv,
                             const float* __restrict__ b,
                             int n_nodes, int F) {
  int idx = blockIdx.x * blockDim.x + threadIdx.x;
  if (idx >= n_nodes * F) return;
  int   i  = idx / F;
  int   f  = idx - i * F;
  float di = dinv[i];
  float v  = agg[idx] + h[idx] * di * di + b[f];
  agg[idx] = v > 0.0f ? v : 0.0f;
}

// ---------------------------------------------------------------------------
extern "C" void kernel_launch(void* const* d_in, const int* in_sizes, int n_in,
                              void* d_out, int out_size, void* d_ws, size_t ws_size,
                              hipStream_t stream) {
  const float* x   = (const float*)d_in[0];
  const int*   ei  = (const int*)  d_in[1];
  const float* W0  = (const float*)d_in[2];
  const float* b0  = (const float*)d_in[3];
  const float* W1  = (const float*)d_in[4];
  const float* b1  = (const float*)d_in[5];
  const float* W2  = (const float*)d_in[6];
  const float* b2  = (const float*)d_in[7];
  const float* fcW = (const float*)d_in[8];
  const float* fcb = (const float*)d_in[9];
  float*       out = (float*)d_out;

  const int* src = ei;              // edge_index[0, :]
  const int* dst = ei + GCN_EDGES;  // edge_index[1, :]

  // Workspace layout: dinv | h | agg | packed bf16 weights  (~38.8 MB)
  float* dinv = (float*)d_ws;
  float* h    = dinv + 50176;                       // padded past 50000
  float* agg  = h + (size_t)GCN_NODES * GCN_HID;
  __bf16* W0p  = (__bf16*)(agg + (size_t)GCN_NODES * GCN_HID);
  __bf16* W1p  = W0p + GCN_INDIM * GCN_HID;         // 49152
  __bf16* W2p  = W1p + GCN_HID * GCN_HID;           // 9216
  __bf16* fcWp = W2p + GCN_HID * GCN_HID;           // 9216 (+6144 for fcWp)

  const int nhid = GCN_NODES * GCN_HID;

  // --- one-time weight pre-pack into WMMA fragment layout (bf16) ---
  gcn_pack_w<<<(GCN_INDIM * GCN_HID + 255) / 256, 256, 0, stream>>>(W0, W0p, GCN_INDIM, GCN_HID);
  gcn_pack_w<<<(GCN_HID  * GCN_HID + 255) / 256, 256, 0, stream>>>(W1, W1p, GCN_HID, GCN_HID);
  gcn_pack_w<<<(GCN_HID  * GCN_HID + 255) / 256, 256, 0, stream>>>(W2, W2p, GCN_HID, GCN_HID);
  gcn_pack_w<<<(GCN_HID  * GCN_OUT + 255) / 256, 256, 0, stream>>>(fcW, fcWp, GCN_HID, GCN_OUT);

  // --- symmetric-normalization degrees: dinv = rsqrt(in_deg + 1) ---
  gcn_zero_f32  <<<(GCN_NODES + 255) / 256, 256, 0, stream>>>(dinv, GCN_NODES);
  gcn_deg_count <<<(GCN_EDGES + 255) / 256, 256, 0, stream>>>(dst, dinv, GCN_EDGES);
  gcn_make_dinv <<<(GCN_NODES + 255) / 256, 256, 0, stream>>>(dinv, GCN_NODES);

  const int gemm_blocks = (GCN_MT * 32 + 255) / 256;   // one wave per M-strip
  const int scat_blocks = (GCN_EDGES * 32 + 255) / 256;

  // ---- layer 0: K=512 ----
  gcn_gemm_wmma<GCN_INDIM, GCN_HID / 16><<<gemm_blocks, 256, 0, stream>>>(
      x, W0p, nullptr, h, GCN_MT);
  gcn_zero_f32<<<(nhid + 255) / 256, 256, 0, stream>>>(agg, nhid);
  gcn_scatter_edges<<<scat_blocks, 256, 0, stream>>>(src, dst, dinv, h, agg, GCN_EDGES);
  gcn_finalize<<<(nhid + 255) / 256, 256, 0, stream>>>(agg, h, dinv, b0, GCN_NODES, GCN_HID);

  // ---- layer 1: K=96 (reads activation in agg, writes h) ----
  gcn_gemm_wmma<GCN_HID, GCN_HID / 16><<<gemm_blocks, 256, 0, stream>>>(
      agg, W1p, nullptr, h, GCN_MT);
  gcn_zero_f32<<<(nhid + 255) / 256, 256, 0, stream>>>(agg, nhid);
  gcn_scatter_edges<<<scat_blocks, 256, 0, stream>>>(src, dst, dinv, h, agg, GCN_EDGES);
  gcn_finalize<<<(nhid + 255) / 256, 256, 0, stream>>>(agg, h, dinv, b1, GCN_NODES, GCN_HID);

  // ---- layer 2: K=96 ----
  gcn_gemm_wmma<GCN_HID, GCN_HID / 16><<<gemm_blocks, 256, 0, stream>>>(
      agg, W2p, nullptr, h, GCN_MT);
  gcn_zero_f32<<<(nhid + 255) / 256, 256, 0, stream>>>(agg, nhid);
  gcn_scatter_edges<<<scat_blocks, 256, 0, stream>>>(src, dst, dinv, h, agg, GCN_EDGES);
  gcn_finalize<<<(nhid + 255) / 256, 256, 0, stream>>>(agg, h, dinv, b2, GCN_NODES, GCN_HID);

  // ---- FC head: out = act @ fcW + fcb (bias fused in GEMM epilogue) ----
  gcn_gemm_wmma<GCN_HID, GCN_OUT / 16><<<gemm_blocks, 256, 0, stream>>>(
      agg, fcWp, fcb, out, GCN_MT);
}
